// MixtureOfExperts_48043504173369
// MI455X (gfx1250) — matrix operand order repaired and checked by reading, not compiled, not run on hardware
//
#include <hip/hip_runtime.h>

#define NTOK 2048
#define CDIM 1024
#define FDIM 4096
#define NEXP 8
#define MTOK 32   // tokens per FFN block (2 WMMA M-tiles)

typedef __bf16 bf16_t;
typedef __attribute__((ext_vector_type(16))) __bf16 v16bf;
typedef __attribute__((ext_vector_type(8)))  __bf16 v8bf;
typedef __attribute__((ext_vector_type(8)))  float  v8f;

__device__ __forceinline__ v16bf cat16(v8bf lo, v8bf hi) {
  return __builtin_shufflevector(lo, hi, 0,1,2,3,4,5,6,7,8,9,10,11,12,13,14,15);
}

// A-matrix fragment (16x32 bf16, M x K) from an LDS row.
// Lane l holds row M=l&15; halves: K = (h>>3)*16 + (l>>4)*8 + (h&7) -> two contiguous
// 16B groups at k0 + hi*8 and k0 + 16 + hi*8.
__device__ __forceinline__ v16bf load_a_lds(const bf16_t* rowptr, int k0, int hi) {
  const v8bf* p0 = (const v8bf*)(rowptr + k0 + hi * 8);
  const v8bf* p1 = (const v8bf*)(rowptr + k0 + 16 + hi * 8);
  return cat16(*p0, *p1);
}

// B-matrix fragment (32x16, K x N) from global fp32 W with leading dim ldw.
// Lane l holds col N = n; halves: K = (h>>3)*16 + (l>>4)*8 + (h&7). fp32 -> bf16 in-register.
__device__ __forceinline__ v16bf load_b_f32(const float* W, int ldw, int k0, int n, int hi) {
  v16bf b;
#pragma unroll
  for (int h = 0; h < 16; ++h) {
    int K = (h >> 3) * 16 + hi * 8 + (h & 7);
    b[h] = (bf16_t)W[(size_t)(k0 + K) * ldw + n];
  }
  return b;
}

__device__ __forceinline__ float silu_f(float g) {
  return g * __builtin_amdgcn_rcpf(1.0f + __expf(-g));
}

// ---------------- kernel 1: zero output + counters ----------------
__global__ void moe_zero(float* out, int* counts) {
  int i = blockIdx.x * blockDim.x + threadIdx.x;
  if (i < NTOK * CDIM) out[i] = 0.0f;
  if (i < NEXP) counts[i] = 0;
}

// ---------------- kernel 2: router (one wave per token) ----------------
__global__ __launch_bounds__(256) void moe_router(const float* __restrict__ xf,
                                                  const float* __restrict__ rw,
                                                  int* gi0, int* gi1, float* gp0, float* gp1,
                                                  float* probs) {
  int wid = (blockIdx.x * blockDim.x + threadIdx.x) >> 5;  // token
  int lane = threadIdx.x & 31;
  if (wid >= NTOK) return;
  const float* xrow = xf + (size_t)wid * CDIM;
  float acc[NEXP] = {0.f, 0.f, 0.f, 0.f, 0.f, 0.f, 0.f, 0.f};
  for (int c = lane; c < CDIM; c += 32) {
    float xv = xrow[c];
#pragma unroll
    for (int e = 0; e < NEXP; ++e) acc[e] += xv * rw[e * CDIM + c];
  }
#pragma unroll
  for (int e = 0; e < NEXP; ++e) {
#pragma unroll
    for (int off = 16; off >= 1; off >>= 1) acc[e] += __shfl_xor(acc[e], off, 32);
  }
  if (lane == 0) {
    int i0 = 0;
#pragma unroll
    for (int e = 1; e < NEXP; ++e) if (acc[e] > acc[i0]) i0 = e;
    int i1 = (i0 == 0) ? 1 : 0;
#pragma unroll
    for (int e = 0; e < NEXP; ++e) {
      if (e != i0 && acc[e] > acc[i1]) i1 = e;
    }
    float p0 = 1.0f / (1.0f + __expf(acc[i1] - acc[i0]));
    gi0[wid] = i0; gi1[wid] = i1; gp0[wid] = p0; gp1[wid] = 1.0f - p0;
    float m = acc[0];
#pragma unroll
    for (int e = 1; e < NEXP; ++e) m = fmaxf(m, acc[e]);
    float s = 0.0f; float ex[NEXP];
#pragma unroll
    for (int e = 0; e < NEXP; ++e) { ex[e] = __expf(acc[e] - m); s += ex[e]; }
    float rs = __builtin_amdgcn_rcpf(s);
#pragma unroll
    for (int e = 0; e < NEXP; ++e) probs[wid * NEXP + e] = ex[e] * rs;
  }
}

// ---------------- kernel 3: deterministic per-expert gather (wave scan) ----------------
__global__ void moe_gather(const int* gi0, const int* gi1, const float* gp0, const float* gp1,
                           int* idx, float* wgt, int* counts) {
  int e = blockIdx.x;  // one wave per expert
  int lane = threadIdx.x;
  int base = 0;
  for (int t0 = 0; t0 < NTOK; t0 += 32) {
    int t = t0 + lane;
    int i0 = gi0[t], i1 = gi1[t];
    float w = (i0 == e) ? gp0[t] : ((i1 == e) ? gp1[t] : 0.0f);
    int f = (i0 == e) || (i1 == e);
    int v = f;
#pragma unroll
    for (int off = 1; off < 32; off <<= 1) {
      int tmp = __shfl_up(v, off, 32);
      if (lane >= off) v += tmp;
    }
    int total = __shfl(v, 31, 32);
    int pre = v - f;
    if (f) { idx[e * NTOK + base + pre] = t; wgt[e * NTOK + base + pre] = w; }
    base += total;
  }
  if (lane == 0) counts[e] = base;
}

// ---------------- kernel 4: aux loss (deterministic tree reduce) ----------------
__global__ void moe_aux(const float* __restrict__ probs, float* out_aux) {
  __shared__ float red[256][NEXP];
  int tid = threadIdx.x;
  float acc[NEXP] = {0.f, 0.f, 0.f, 0.f, 0.f, 0.f, 0.f, 0.f};
  for (int t = tid; t < NTOK; t += 256) {
#pragma unroll
    for (int e = 0; e < NEXP; ++e) acc[e] += probs[t * NEXP + e];
  }
#pragma unroll
  for (int e = 0; e < NEXP; ++e) red[tid][e] = acc[e];
  __syncthreads();
  for (int s = 128; s > 0; s >>= 1) {
    if (tid < s) {
#pragma unroll
      for (int e = 0; e < NEXP; ++e) red[tid][e] += red[tid + s][e];
    }
    __syncthreads();
  }
  if (tid == 0) {
    float sum = 0.0f;
#pragma unroll
    for (int e = 0; e < NEXP; ++e) {
      float m = red[0][e] / (float)NTOK;
      sum += m * m;
    }
    *out_aux = 0.01f * (float)NEXP * sum;
  }
}

// ---------------- kernel 5: grouped SwiGLU FFN via bf16 WMMA ----------------
// Block = (expert e, 32-token tile), 256 threads = 8 waves.
// Phase 1 (per 64-wide F chunk): waves 0-3 gate, 4-7 up; each wave owns one N-tile
//   and both M-tiles, so each B fragment feeds 2 WMMAs (8 VMEM/WMMA).
// Phase 2: h = silu(gate)*up*combine_weight -> bf16 LDS (branch-free, b128 LDS ops).
// Phase 3: down-GEMM; each wave owns 128 output columns x 32 rows (16 f32 tiles in
//   registers); B fragments shared across the two M-tiles.
__global__ __launch_bounds__(256) void moe_ffn(const float* __restrict__ xf,
                                               const float* __restrict__ w1,
                                               const float* __restrict__ w3,
                                               const float* __restrict__ w2,
                                               const int* __restrict__ counts,
                                               const int* __restrict__ idx,
                                               const float* __restrict__ wgt,
                                               float* __restrict__ out) {
  const int e = blockIdx.y;
  const int tile = blockIdx.x;
  const int cnt = counts[e];
  if (tile * MTOK >= cnt) return;  // block-uniform

  __shared__ __align__(16) bf16_t xs[MTOK][CDIM];   // 64 KB
  __shared__ __align__(16) float  gbuf[MTOK][64];   // 8 KB
  __shared__ __align__(16) float  ubuf[MTOK][64];   // 8 KB
  __shared__ __align__(16) bf16_t hbuf[MTOK][64];   // 4 KB
  __shared__ int   tok[MTOK];
  __shared__ float twt[MTOK];

  const int tid = threadIdx.x;
  const int lane = tid & 31;
  const int wv = tid >> 5;
  const int hi = lane >> 4;
  const int nn = lane & 15;

  if (tid < MTOK) {
    int s = tile * MTOK + tid;
    if (s < cnt) { tok[tid] = idx[e * NTOK + s]; twt[tid] = wgt[e * NTOK + s]; }
    else         { tok[tid] = 0;                 twt[tid] = 0.0f; }
  }
  __syncthreads();

  // stage x tile -> bf16 LDS (float4 loads, pk converts)
  for (int i = tid; i < MTOK * (CDIM / 4); i += 256) {
    int row = i >> 8, c4 = i & 255;
    float4 v = ((const float4*)(xf + (size_t)tok[row] * CDIM))[c4];
    bf16_t* dst = &xs[row][c4 * 4];
    dst[0] = (bf16_t)v.x; dst[1] = (bf16_t)v.y;
    dst[2] = (bf16_t)v.z; dst[3] = (bf16_t)v.w;
  }
  __syncthreads();

  const float* wu  = ((wv < 4) ? w1 : w3) + (size_t)e * CDIM * FDIM;  // (C,F)
  const float* w2e = w2 + (size_t)e * FDIM * CDIM;                    // (F,C)
  const int ntile = wv & 3;

  // phase-2 static assignment: one contiguous 8-col strip of one row per thread
  const int p2row = tid >> 3;
  const int p2c0  = (tid & 7) * 8;

  v8f oacc[16];
#pragma unroll
  for (int i = 0; i < 16; ++i) oacc[i] = (v8f){0.f, 0.f, 0.f, 0.f, 0.f, 0.f, 0.f, 0.f};

  for (int f0 = 0; f0 < FDIM; f0 += 64) {
    // ---- phase 1: gate/up (32 x 16 tile pair), K-loop over C ----
    int ncol = f0 + ntile * 16 + nn;
    v8f acc0 = (v8f){0.f, 0.f, 0.f, 0.f, 0.f, 0.f, 0.f, 0.f};
    v8f acc1 = acc0;
#pragma unroll 2
    for (int k0 = 0; k0 < CDIM; k0 += 32) {
      v16bf b  = load_b_f32(wu, FDIM, k0, ncol, hi);
      v16bf a0 = load_a_lds(&xs[nn][0], k0, hi);
      v16bf a1 = load_a_lds(&xs[16 + nn][0], k0, hi);
      acc0 = __builtin_amdgcn_wmma_f32_16x16x32_bf16(false, a0, false, b, (short)0, acc0,
                                                     false, false);
      acc1 = __builtin_amdgcn_wmma_f32_16x16x32_bf16(false, a1, false, b, (short)0, acc1,
                                                     false, false);
    }
    float* buf = (wv < 4) ? &gbuf[0][0] : &ubuf[0][0];
#pragma unroll
    for (int r = 0; r < 8; ++r) {
      int M = r + 8 * hi;
      buf[M * 64 + ntile * 16 + nn] = acc0[r];
      buf[(16 + M) * 64 + ntile * 16 + nn] = acc1[r];
    }
    __syncthreads();

    // ---- phase 2: h = silu(gate)*up * combine_weight, bf16 (branch-free) ----
    {
      float tw = twt[p2row];
      const float4* gp = (const float4*)&gbuf[p2row][p2c0];
      const float4* up = (const float4*)&ubuf[p2row][p2c0];
      float4 g0 = gp[0], g1 = gp[1];
      float4 u0 = up[0], u1 = up[1];
      float gg[8] = {g0.x, g0.y, g0.z, g0.w, g1.x, g1.y, g1.z, g1.w};
      float uu[8] = {u0.x, u0.y, u0.z, u0.w, u1.x, u1.y, u1.z, u1.w};
      v8bf hv;
#pragma unroll
      for (int j = 0; j < 8; ++j) hv[j] = (bf16_t)(silu_f(gg[j]) * uu[j] * tw);
      *(v8bf*)&hbuf[p2row][p2c0] = hv;
    }
    __syncthreads();

    // ---- phase 3: down-GEMM: out(32x1024) += h(32x64) @ w2[f0:f0+64, :] ----
    v16bf a00 = load_a_lds(&hbuf[nn][0], 0, hi);
    v16bf a01 = load_a_lds(&hbuf[nn][0], 32, hi);
    v16bf a10 = load_a_lds(&hbuf[16 + nn][0], 0, hi);
    v16bf a11 = load_a_lds(&hbuf[16 + nn][0], 32, hi);
#pragma unroll
    for (int nt = 0; nt < 8; ++nt) {
      int n = wv * 128 + nt * 16 + nn;
      v16bf b0 = load_b_f32(w2e, CDIM, f0, n, hi);
      v16bf b1 = load_b_f32(w2e, CDIM, f0 + 32, n, hi);
      oacc[nt] = __builtin_amdgcn_wmma_f32_16x16x32_bf16(false, a00, false, b0, (short)0,
                                                         oacc[nt], false, false);
      oacc[nt] = __builtin_amdgcn_wmma_f32_16x16x32_bf16(false, a01, false, b1, (short)0,
                                                         oacc[nt], false, false);
      oacc[8 + nt] = __builtin_amdgcn_wmma_f32_16x16x32_bf16(false, a10, false, b0, (short)0,
                                                             oacc[8 + nt], false, false);
      oacc[8 + nt] = __builtin_amdgcn_wmma_f32_16x16x32_bf16(false, a11, false, b1, (short)0,
                                                             oacc[8 + nt], false, false);
    }
    __syncthreads();  // protect g/u/h buffers before next chunk
  }

  // scatter-accumulate (<=2 experts per token: commutative f32 adds -> deterministic)
#pragma unroll
  for (int mt = 0; mt < 2; ++mt) {
#pragma unroll
    for (int nt = 0; nt < 8; ++nt) {
      int n = wv * 128 + nt * 16 + nn;
#pragma unroll
      for (int r = 0; r < 8; ++r) {
        int M = mt * 16 + r + 8 * hi;
        atomicAdd(&out[(size_t)tok[M] * CDIM + n], oacc[mt * 8 + nt][r]);
      }
    }
  }
}

extern "C" void kernel_launch(void* const* d_in, const int* in_sizes, int n_in,
                              void* d_out, int out_size, void* d_ws, size_t ws_size,
                              hipStream_t stream) {
  const float* x  = (const float*)d_in[0];  // (2,1024,1024)
  const float* rw = (const float*)d_in[1];  // (8,1024)
  const float* w1 = (const float*)d_in[2];  // (8,1024,4096)
  const float* w3 = (const float*)d_in[3];  // (8,1024,4096)
  const float* w2 = (const float*)d_in[4];  // (8,4096,1024)
  float* out = (float*)d_out;               // 2048*1024 out + 1 aux

  char* ws = (char*)d_ws;
  int*   counts = (int*)ws;
  int*   gi0    = (int*)(ws + 64);
  int*   gi1    = gi0 + NTOK;
  float* gp0    = (float*)(gi1 + NTOK);
  float* gp1    = gp0 + NTOK;
  float* probs  = gp1 + NTOK;
  int*   idx    = (int*)(probs + NTOK * NEXP);
  float* wgt    = (float*)(idx + NEXP * NTOK);

  moe_zero<<<(NTOK * CDIM + 255) / 256, 256, 0, stream>>>(out, counts);
  moe_router<<<(NTOK * 32 + 255) / 256, 256, 0, stream>>>(x, rw, gi0, gi1, gp0, gp1, probs);
  moe_gather<<<NEXP, 32, 0, stream>>>(gi0, gi1, gp0, gp1, idx, wgt, counts);
  moe_aux<<<1, 256, 0, stream>>>(probs, out + (size_t)NTOK * CDIM);
  moe_ffn<<<dim3(NTOK / MTOK, NEXP), 256, 0, stream>>>(x, w1, w3, w2, counts, idx, wgt, out);
}
